// dygcc_cvx_Block_79224966742733
// MI455X (gfx1250) — matrix operand order repaired and checked by reading, not compiled, not run on hardware
//
#include <hip/hip_runtime.h>
#include <cstdint>
#include <cstddef>

// Shapes (fixed by the reference)
//   B=16, DIM=512, H=W=32, CH=256, RED=64, L=32, HID=2048, NTOK=16384

typedef __attribute__((ext_vector_type(16))) _Float16 v16h;
typedef __attribute__((ext_vector_type(8)))  _Float16 v8h;
typedef __attribute__((ext_vector_type(8)))  float    v8f;
typedef __attribute__((ext_vector_type(2)))  float    v2f;

__device__ __forceinline__ float wave_sum32(float v) {
#pragma unroll
  for (int o = 16; o > 0; o >>= 1) v += __shfl_xor(v, o, 32);
  return v;
}

// ---------------------------------------------------------------------------
// K0: convert & transpose MLP weights to f16 (K-contiguous per output column)
//     w1t[n][k] = pw1_w[k][n]  (n<2048,k<512) ; w2t[n][k] = pw2_w[k][n]
// ---------------------------------------------------------------------------
__global__ __launch_bounds__(256) void k0_cvt(const float* __restrict__ w1,
                                              const float* __restrict__ w2,
                                              _Float16* __restrict__ w1t,
                                              _Float16* __restrict__ w2t) {
  int idx = blockIdx.x * 256 + threadIdx.x;      // 0 .. 1048575
  {
    int n = idx >> 9, k = idx & 511;             // w1t[n*512+k]
    w1t[idx] = (_Float16)w1[k * 2048 + n];
  }
  {
    int n = idx >> 11, k = idx & 2047;           // w2t[n*2048+k]
    w2t[idx] = (_Float16)w2[k * 512 + n];
  }
}

// ---------------------------------------------------------------------------
// K1: gin[br][b][cc][l] = info + glob  (one wave per (b,c))
// ---------------------------------------------------------------------------
__global__ __launch_bounds__(256) void k1_gin(const float* __restrict__ x,
                                              const float* __restrict__ pe_h,
                                              const float* __restrict__ pe_w,
                                              float* __restrict__ gin) {
  int lane = threadIdx.x & 31, wid = threadIdx.x >> 5;
  int gw = blockIdx.x * 8 + wid;                 // 0 .. 8191
  int b = gw >> 9, c = gw & 511;
  int br = c >> 8, cc = c & 255;
  const float* xp = x + ((size_t)b * 512 + c) * 1024;
  const float* pe = br ? pe_w : pe_h;
  float s = 0.f;
  if (br == 0) {                                 // lane = h, reduce over w
#pragma unroll 4
    for (int w = 0; w < 32; ++w) s += xp[lane * 32 + w];
  } else {                                       // lane = w, reduce over h
#pragma unroll 4
    for (int h = 0; h < 32; ++h) s += xp[h * 32 + lane];
  }
  float tot = wave_sum32(s);                     // sum over all 1024
  float pv  = pe[cc * 32 + lane];
  float pm  = wave_sum32(pv);                    // sum of pe over axis
  float g = s * (1.f / 32.f) + pv + tot * (1.f / 1024.f) + pm * (1.f / 32.f);
  gin[((size_t)(br * 16 + b) * 256 + cc) * 32 + lane] = g;
}

// ---------------------------------------------------------------------------
// K2: filter generator (_gen): conv1 + BN + hardswish + conv2 + filternorm
//     one block (256 thr) per (br,b); outputs wcoef[br][b][c][l]
// ---------------------------------------------------------------------------
__global__ __launch_bounds__(256) void k2_gen(
    const float* __restrict__ gin,
    const float* __restrict__ g1w_h, const float* __restrict__ g1w_w,
    const float* __restrict__ bng_h, const float* __restrict__ bng_w,
    const float* __restrict__ bnb_h, const float* __restrict__ bnb_w,
    const float* __restrict__ bnrm_h, const float* __restrict__ bnrm_w,
    const float* __restrict__ bnrv_h, const float* __restrict__ bnrv_w,
    const float* __restrict__ g2w_h, const float* __restrict__ g2w_w,
    const float* __restrict__ fns_h, const float* __restrict__ fns_w,
    const float* __restrict__ fnm_h, const float* __restrict__ fnm_w,
    float* __restrict__ wcoef) {
  __shared__ float gs[8192];     // gin tile 256x32
  __shared__ float h1s[2048];    // 64x32 hidden
  __shared__ float rsum[256];
  __shared__ float rsq[256];
  const int tid = threadIdx.x;
  const int id = blockIdx.x;     // br*16 + b
  const int br = id >> 4;
  const float* w1   = br ? g1w_w  : g1w_h;
  const float* w2   = br ? g2w_w  : g2w_h;
  const float* bng  = br ? bng_w  : bng_h;
  const float* bnb  = br ? bnb_w  : bnb_h;
  const float* bnrm = br ? bnrm_w : bnrm_h;
  const float* bnrv = br ? bnrv_w : bnrv_h;
  const float* fns  = br ? fns_w  : fns_h;
  const float* fnm  = br ? fnm_w  : fnm_h;

  const float* gp = gin + (size_t)id * 8192;
  for (int t = tid; t < 8192; t += 256) gs[t] = gp[t];
  __syncthreads();

  // conv1 (CH->RED, k=3, pad 1): thread -> (r, 8 consecutive l)
  {
    int r = tid & 63, l0 = (tid >> 6) * 8;
    float a[8];
#pragma unroll
    for (int u = 0; u < 8; ++u) a[u] = 0.f;
    for (int c = 0; c < 256; ++c) {
      const float* wr = w1 + (r * 256 + c) * 3;
      float k0 = wr[0], k1 = wr[1], k2 = wr[2];
      const float* row = gs + c * 32;
      float g[10];
#pragma unroll
      for (int j = 0; j < 10; ++j) {
        int l = l0 - 1 + j;
        g[j] = (l >= 0 && l < 32) ? row[l] : 0.f;
      }
#pragma unroll
      for (int u = 0; u < 8; ++u)
        a[u] = fmaf(k0, g[u], fmaf(k1, g[u + 1], fmaf(k2, g[u + 2], a[u])));
    }
    float sc = bng[r] * rsqrtf(bnrv[r] + 1e-5f);
    float sh = bnb[r] - bnrm[r] * sc;
#pragma unroll
    for (int u = 0; u < 8; ++u) {
      float h = a[u] * sc + sh;                        // BN
      float hs = h * fminf(fmaxf(h + 3.f, 0.f), 6.f) * (1.f / 6.f); // hardswish
      h1s[r * 32 + l0 + u] = hs;
    }
  }
  __syncthreads();

  // conv2 (RED->CH, k=3, pad 1): thread -> channel c, all 32 l in registers
  int c = tid;
  float o[32];
#pragma unroll
  for (int l = 0; l < 32; ++l) o[l] = 0.f;
  for (int r = 0; r < 64; ++r) {
    const float* wr = w2 + (c * 64 + r) * 3;
    float k0 = wr[0], k1 = wr[1], k2 = wr[2];
    const float* row = h1s + r * 32;
    float prev = 0.f, cur = row[0];
#pragma unroll
    for (int l = 0; l < 32; ++l) {
      float nxt = (l < 31) ? row[l + 1] : 0.f;
      o[l] = fmaf(k0, prev, fmaf(k1, cur, fmaf(k2, nxt, o[l])));
      prev = cur; cur = nxt;
    }
  }
  // filter-norm over (c,l) = 8192 values
  float ls = 0.f, lq = 0.f;
#pragma unroll
  for (int l = 0; l < 32; ++l) { ls += o[l]; lq += o[l] * o[l]; }
  rsum[tid] = ls; rsq[tid] = lq;
  __syncthreads();
  for (int st = 128; st > 0; st >>= 1) {
    if (tid < st) { rsum[tid] += rsum[tid + st]; rsq[tid] += rsq[tid + st]; }
    __syncthreads();
  }
  float u  = rsum[0] * (1.f / 8192.f);
  float s2 = rsq[0] * (1.f / 8192.f) - u * u;
  float ri = rsqrtf(s2 + 1e-6f);
  float* wout = wcoef + (size_t)id * 8192 + c * 32;
#pragma unroll
  for (int l = 0; l < 32; ++l)
    wout[l] = (o[l] - u) * ri * fns[c * 32 + l] + fnm[c * 32 + l];
}

// ---------------------------------------------------------------------------
// K3: circulant 32x32x32 GEMM per (b,c) with f32 WMMA 16x16x4.
//     br0: D = Circ(w) * Xp ; br1: D = Xp * Circ'(w).  One wave per (b,c).
//     A and B^T staged in LDS with stride-34 rows (bank-conflict free,
//     8B-aligned pairs for the per-lane (k,k+1) fragment loads).
// ---------------------------------------------------------------------------
__global__ __launch_bounds__(128) void k3_circ(const float* __restrict__ x,
                                               const float* __restrict__ wcoef,
                                               const float* __restrict__ pe_h,
                                               const float* __restrict__ pe_w,
                                               const float* __restrict__ bias_h,
                                               const float* __restrict__ bias_w,
                                               float* __restrict__ ytok) {
  __shared__ float sh[4 * 2208];        // per wave: At[32*34] Bt[32*34] wl[32]
  int lane = threadIdx.x & 31, wid = threadIdx.x >> 5;
  int gw = blockIdx.x * 4 + wid;        // 0 .. 8191
  int b = gw >> 9, c = gw & 511;
  int br = c >> 8, cc = c & 255;
  float* At = sh + wid * 2208;
  float* Bt = At + 1088;
  float* wl = Bt + 1088;
  wl[lane] = wcoef[((size_t)(br * 16 + b) * 256 + cc) * 32 + lane];
  __syncthreads();
  const float* xp = x + ((size_t)b * 512 + c) * 1024;
  if (br == 0) {
    // B[k][n] = x[k][n] + pe_h[cc][k]  -> Bt[n][k]
#pragma unroll 4
    for (int k = 0; k < 32; ++k)
      Bt[lane * 34 + k] = xp[k * 32 + lane] + pe_h[cc * 32 + k];
    // A[i][j] = w[(j-i) mod 32]
#pragma unroll 4
    for (int i = 0; i < 32; ++i)
      At[i * 34 + lane] = wl[(lane - i) & 31];
  } else {
    // A[i][j] = x[i][j] + pe_w[cc][j]
#pragma unroll 4
    for (int i = 0; i < 32; ++i)
      At[i * 34 + lane] = xp[i * 32 + lane] + pe_w[cc * 32 + lane];
    // B[k][n] = w[(n-k) mod 32] -> Bt[n][k]
#pragma unroll 4
    for (int k = 0; k < 32; ++k)
      Bt[lane * 34 + k] = wl[(lane - k) & 31];
  }
  __syncthreads();

  int lm = lane & 15, hi = lane >> 4;
  v8f acc[2][2] = {};
#pragma unroll
  for (int kk = 0; kk < 8; ++kk) {
    int ko = kk * 4 + hi * 2;           // lane<16 -> k,k+1 ; lane>=16 -> k+2,k+3
    v2f af[2], bf[2];
#pragma unroll
    for (int mi = 0; mi < 2; ++mi) {
      const float* p = At + (mi * 16 + lm) * 34 + ko;
      af[mi][0] = p[0]; af[mi][1] = p[1];
    }
#pragma unroll
    for (int ni = 0; ni < 2; ++ni) {
      const float* p = Bt + (ni * 16 + lm) * 34 + ko;
      bf[ni][0] = p[0]; bf[ni][1] = p[1];
    }
#pragma unroll
    for (int mi = 0; mi < 2; ++mi)
#pragma unroll
      for (int ni = 0; ni < 2; ++ni)
        acc[mi][ni] = __builtin_amdgcn_wmma_f32_16x16x4_f32(
            false, af[mi], false, bf[ni], (short)0, acc[mi][ni], false, false);
  }

  float bias = (br ? bias_w : bias_h)[cc];
  float* yt = ytok + (size_t)b * 1024 * 512 + (size_t)(br * 256 + cc);
#pragma unroll
  for (int mi = 0; mi < 2; ++mi)
#pragma unroll
    for (int ni = 0; ni < 2; ++ni)
#pragma unroll
      for (int v = 0; v < 8; ++v) {
        int M = mi * 16 + hi * 8 + v;   // D-matrix VGPR layout
        int N = ni * 16 + lm;
        yt[((size_t)M * 32 + N) * 512] = acc[mi][ni][v] + bias;
      }
}

// ---------------------------------------------------------------------------
// K4: LayerNorm over 512, emit f16 tokens.  One wave per token.
// ---------------------------------------------------------------------------
__global__ __launch_bounds__(256) void k4_ln(const float* __restrict__ ytok,
                                             const float* __restrict__ lnw,
                                             const float* __restrict__ lnb,
                                             _Float16* __restrict__ yn) {
  int lane = threadIdx.x & 31, wid = threadIdx.x >> 5;
  int t = blockIdx.x * 8 + wid;
  const float* row = ytok + (size_t)t * 512;
  float vals[16];
  float s = 0.f;
#pragma unroll
  for (int j = 0; j < 16; ++j) { vals[j] = row[j * 32 + lane]; s += vals[j]; }
  float mu = wave_sum32(s) * (1.f / 512.f);
  float q = 0.f;
#pragma unroll
  for (int j = 0; j < 16; ++j) { float d = vals[j] - mu; q += d * d; }
  float var = wave_sum32(q) * (1.f / 512.f);
  float rs = rsqrtf(var + 1e-6f);
  _Float16* orow = yn + (size_t)t * 512;
#pragma unroll
  for (int j = 0; j < 16; ++j) {
    int ch = j * 32 + lane;
    orow[ch] = (_Float16)((vals[j] - mu) * rs * lnw[ch] + lnb[ch]);
  }
}

// ---------------------------------------------------------------------------
// K5: GEMM1  (16384x512) x (512x2048) + bias, exact GELU -> hid (f16)
//     f16 WMMA 16x16x32; block = 8 waves, tile 64(M) x 128(N).
// ---------------------------------------------------------------------------
__global__ __launch_bounds__(256) void k5_gemm1(const _Float16* __restrict__ A,
                                                const _Float16* __restrict__ Bt,
                                                const float* __restrict__ bias,
                                                _Float16* __restrict__ hid) {
  int lane = threadIdx.x & 31, wid = threadIdx.x >> 5;
  int lm = lane & 15, hi = lane >> 4;
  int wm = wid & 3, wn = wid >> 2;
  int mbase = blockIdx.y * 64 + wm * 16;
  int nbase = blockIdx.x * 128 + wn * 64;
  const _Float16* arow = A + (size_t)(mbase + lm) * 512;
  v8f acc[4] = {};
  for (int k0 = 0; k0 < 512; k0 += 32) {
    v8h alo = *(const v8h*)(arow + k0 + hi * 16);       // K chunk [16*hi, 16*hi+8)
    v8h ahi = *(const v8h*)(arow + k0 + hi * 16 + 8);   // K chunk [16*hi+8, 16*hi+16)
    v16h av = __builtin_shufflevector(alo, ahi, 0, 1, 2, 3, 4, 5, 6, 7,
                                      8, 9, 10, 11, 12, 13, 14, 15);
#pragma unroll
    for (int ni = 0; ni < 4; ++ni) {
      const _Float16* bcol = Bt + (size_t)(nbase + ni * 16 + lm) * 512 + k0 + hi * 16;
      v8h blo = *(const v8h*)(bcol);
      v8h bhi = *(const v8h*)(bcol + 8);
      v16h bv = __builtin_shufflevector(blo, bhi, 0, 1, 2, 3, 4, 5, 6, 7,
                                        8, 9, 10, 11, 12, 13, 14, 15);
      acc[ni] = __builtin_amdgcn_wmma_f32_16x16x32_f16(
          false, av, false, bv, (short)0, acc[ni], false, false);
    }
  }
#pragma unroll
  for (int ni = 0; ni < 4; ++ni) {
    int n = nbase + ni * 16 + lm;
    float bv = bias[n];
#pragma unroll
    for (int v = 0; v < 8; ++v) {
      int M = mbase + hi * 8 + v;
      float h = acc[ni][v] + bv;
      float g = 0.5f * h * (1.f + erff(h * 0.70710678118654752f)); // exact GELU
      hid[(size_t)M * 2048 + n] = (_Float16)g;
    }
  }
}

// ---------------------------------------------------------------------------
// K6: GEMM2 (16384x2048) x (2048x512) + bias, * gamma, + residual -> out f32
// ---------------------------------------------------------------------------
__global__ __launch_bounds__(256) void k6_gemm2(const _Float16* __restrict__ A,
                                                const _Float16* __restrict__ Bt,
                                                const float* __restrict__ bias,
                                                const float* __restrict__ gamma,
                                                const float* __restrict__ xin,
                                                float* __restrict__ out) {
  int lane = threadIdx.x & 31, wid = threadIdx.x >> 5;
  int lm = lane & 15, hi = lane >> 4;
  int wm = wid & 3, wn = wid >> 2;
  int mbase = blockIdx.y * 64 + wm * 16;
  int nbase = blockIdx.x * 128 + wn * 64;
  const _Float16* arow = A + (size_t)(mbase + lm) * 2048;
  v8f acc[4] = {};
  for (int k0 = 0; k0 < 2048; k0 += 32) {
    v8h alo = *(const v8h*)(arow + k0 + hi * 16);
    v8h ahi = *(const v8h*)(arow + k0 + hi * 16 + 8);
    v16h av = __builtin_shufflevector(alo, ahi, 0, 1, 2, 3, 4, 5, 6, 7,
                                      8, 9, 10, 11, 12, 13, 14, 15);
#pragma unroll
    for (int ni = 0; ni < 4; ++ni) {
      const _Float16* bcol = Bt + (size_t)(nbase + ni * 16 + lm) * 2048 + k0 + hi * 16;
      v8h blo = *(const v8h*)(bcol);
      v8h bhi = *(const v8h*)(bcol + 8);
      v16h bv = __builtin_shufflevector(blo, bhi, 0, 1, 2, 3, 4, 5, 6, 7,
                                        8, 9, 10, 11, 12, 13, 14, 15);
      acc[ni] = __builtin_amdgcn_wmma_f32_16x16x32_f16(
          false, av, false, bv, (short)0, acc[ni], false, false);
    }
  }
#pragma unroll
  for (int ni = 0; ni < 4; ++ni) {
    int n = nbase + ni * 16 + lm;
    float bv = bias[n];
    float gm = gamma[n];
#pragma unroll
    for (int v = 0; v < 8; ++v) {
      int M = mbase + hi * 8 + v;              // token id
      int bb = M >> 10, hw = M & 1023;
      size_t oi = ((size_t)bb * 512 + n) * 1024 + hw;   // (B,DIM,H,W) layout
      out[oi] = xin[oi] + gm * (acc[ni][v] + bv);
    }
  }
}

// ---------------------------------------------------------------------------
extern "C" void kernel_launch(void* const* d_in, const int* in_sizes, int n_in,
                              void* d_out, int out_size, void* d_ws, size_t ws_size,
                              hipStream_t stream) {
  const float* x      = (const float*)d_in[0];
  const float* pe_h   = (const float*)d_in[1];
  const float* g1w_h  = (const float*)d_in[2];
  const float* bng_h  = (const float*)d_in[3];
  const float* bnb_h  = (const float*)d_in[4];
  const float* bnrm_h = (const float*)d_in[5];
  const float* bnrv_h = (const float*)d_in[6];
  const float* g2w_h  = (const float*)d_in[7];
  const float* fns_h  = (const float*)d_in[8];
  const float* fnm_h  = (const float*)d_in[9];
  const float* bias_h = (const float*)d_in[10];
  const float* pe_w   = (const float*)d_in[11];
  const float* g1w_w  = (const float*)d_in[12];
  const float* bng_w  = (const float*)d_in[13];
  const float* bnb_w  = (const float*)d_in[14];
  const float* bnrm_w = (const float*)d_in[15];
  const float* bnrv_w = (const float*)d_in[16];
  const float* g2w_w  = (const float*)d_in[17];
  const float* fns_w  = (const float*)d_in[18];
  const float* fnm_w  = (const float*)d_in[19];
  const float* bias_w = (const float*)d_in[20];
  const float* ln_w   = (const float*)d_in[21];
  const float* ln_b   = (const float*)d_in[22];
  const float* pw1_w  = (const float*)d_in[23];
  const float* pw1_b  = (const float*)d_in[24];
  const float* pw2_w  = (const float*)d_in[25];
  const float* pw2_b  = (const float*)d_in[26];
  const float* gamma  = (const float*)d_in[27];

  char* ws = (char*)d_ws;
  float*    gin   = (float*)(ws + ((size_t)0 << 20));   //  1 MB [2][16][256][32]
  float*    wcoef = (float*)(ws + ((size_t)1 << 20));   //  1 MB [2][16][256][32]
  float*    ytok  = (float*)(ws + ((size_t)2 << 20));   // 32 MB [16][32][32][512]
  _Float16* ynorm = (_Float16*)(ws + ((size_t)34 << 20)); // 16 MB [16384][512]
  _Float16* w1t   = (_Float16*)(ws + ((size_t)50 << 20)); //  2 MB [2048][512]
  _Float16* w2t   = (_Float16*)(ws + ((size_t)52 << 20)); //  2 MB [512][2048]
  _Float16* hid   = (_Float16*)(ws + ((size_t)54 << 20)); // 64 MB [16384][2048]

  k0_cvt<<<4096, 256, 0, stream>>>(pw1_w, pw2_w, w1t, w2t);
  k1_gin<<<1024, 256, 0, stream>>>(x, pe_h, pe_w, gin);
  k2_gen<<<32, 256, 0, stream>>>(gin, g1w_h, g1w_w, bng_h, bng_w, bnb_h, bnb_w,
                                 bnrm_h, bnrm_w, bnrv_h, bnrv_w, g2w_h, g2w_w,
                                 fns_h, fns_w, fnm_h, fnm_w, wcoef);
  k3_circ<<<2048, 128, 0, stream>>>(x, wcoef, pe_h, pe_w, bias_h, bias_w, ytok);
  k4_ln<<<2048, 256, 0, stream>>>(ytok, ln_w, ln_b, ynorm);
  dim3 g5(16, 256);
  k5_gemm1<<<g5, 256, 0, stream>>>(ynorm, w1t, pw1_b, hid);
  dim3 g6(4, 256);
  k6_gemm2<<<g6, 256, 0, stream>>>(hid, w2t, pw2_b, gamma, x, (float*)d_out);
}
